// SelfAttention_57664230916926
// MI455X (gfx1250) — matrix operand order repaired and checked by reading, not compiled, not run on hardware
//
#include <hip/hip_runtime.h>

// ---------------------------------------------------------------------------
// MI455X (gfx1250) fused causal self-attention, compile-only target.
// Pipeline: [K1] QKV GEMM (f32->bf16, v_wmma_f32_16x16x32_bf16)
//           [K2] flash attention per (b, head, 64-row block)
//                K-tiles staged with GLOBAL_LOAD_ASYNC_TO_LDS_B128 (ASYNCcnt)
//           [K3] output projection GEMM (bf16 x f32w -> f32 out),
//                A-tiles staged async as well.
// ---------------------------------------------------------------------------

typedef __attribute__((ext_vector_type(16))) __bf16 v16bf;
typedef __attribute__((ext_vector_type(8)))  float  v8f;

union FragBF {
  v16bf v;
  __bf16 e[16];
  uint4  q[2];
};

static __device__ __forceinline__ v8f wmma_bf16(v16bf a, v16bf b, v8f c) {
  // (neg_a, A, neg_b, B, c_mod, C, reuse_a, reuse_b)
  return __builtin_amdgcn_wmma_f32_16x16x32_bf16(false, a, false, b, (short)0, c,
                                                 false, false);
}

static __device__ __forceinline__ v8f vzero8() {
  v8f z = {0.f, 0.f, 0.f, 0.f, 0.f, 0.f, 0.f, 0.f};
  return z;
}

// ds_swizzle group-of-32: offset = {xor[14:10], or[9:5], and[4:0]}
template <int IMM>
static __device__ __forceinline__ float swz(float x) {
  return __int_as_float(__builtin_amdgcn_ds_swizzle(__float_as_int(x), IMM));
}
static __device__ __forceinline__ float redmax16(float v) {
  v = fmaxf(v, swz<0x041F>(v));   // xor 1
  v = fmaxf(v, swz<0x081F>(v));   // xor 2
  v = fmaxf(v, swz<0x101F>(v));   // xor 4
  v = fmaxf(v, swz<0x201F>(v));   // xor 8
  return v;
}
static __device__ __forceinline__ float redsum16(float v) {
  v += swz<0x041F>(v);
  v += swz<0x081F>(v);
  v += swz<0x101F>(v);
  v += swz<0x201F>(v);
  return v;
}

// Async global->LDS copies (GV mode, saddr=NULL). INST_OFFSET is added to
// BOTH the global and the LDS address, so contiguous chunks on both sides
// can share one LDS-addr VGPR + one global-addr VGPR pair.
static __device__ __forceinline__ void async_copy64(unsigned lds_off,
                                                    unsigned long long gaddr) {
  asm volatile(
      "global_load_async_to_lds_b128 %0, %1, off\n\t"
      "global_load_async_to_lds_b128 %0, %1, off offset:16\n\t"
      "global_load_async_to_lds_b128 %0, %1, off offset:32\n\t"
      "global_load_async_to_lds_b128 %0, %1, off offset:48"
      :: "v"(lds_off), "v"(gaddr) : "memory");
}
static __device__ __forceinline__ void async_copy32(unsigned lds_off,
                                                    unsigned long long gaddr) {
  asm volatile(
      "global_load_async_to_lds_b128 %0, %1, off\n\t"
      "global_load_async_to_lds_b128 %0, %1, off offset:16"
      :: "v"(lds_off), "v"(gaddr) : "memory");
}
static __device__ __forceinline__ void wait_async0() {
  asm volatile("s_wait_asynccnt 0x0" ::: "memory");
}
static __device__ __forceinline__ unsigned lds_addr_of(const void* p) {
  return (unsigned)(unsigned long long)p;  // low 32 bits = LDS byte offset
}

// ---------------------------------------------------------------------------
// Generic WMMA GEMM: C[M,N] = A[M,K] * B[K,N].
// Block tile 128(M) x 64(N); 8 waves, each wave owns a 32x32 sub-tile
// (2x2 WMMA 16x16 tiles). K processed in 32-wide chunks through LDS.
// A staged row-major [128][32] (async when already bf16); B staged TRANSPOSED
// [64][32] so B-fragments read contiguous 16B chunks.
// ---------------------------------------------------------------------------
template <bool A_F32, bool OUT_BF16>
__global__ __launch_bounds__(256)
void wmma_gemm_kernel(const void* __restrict__ Ap, const float* __restrict__ Bp,
                      void* __restrict__ Op, int K, int lda, int ldb, int ldo) {
  __shared__ __align__(16) __bf16 As[128][40];
  __shared__ __align__(16) __bf16 BsT[64][40];

  const int tid  = threadIdx.x;
  const int lane = tid & 31;
  const int wv   = tid >> 5;
  const int l15  = lane & 15;
  const int half = lane >> 4;
  const int wm   = (wv & 3) * 32;
  const int wn   = (wv >> 2) * 32;
  const int mBase = blockIdx.x * 128;
  const int nBase = blockIdx.y * 64;

  v8f acc[2][2] = {{vzero8(), vzero8()}, {vzero8(), vzero8()}};

  const int mrow = tid >> 1;          // 0..127
  const int koff = (tid & 1) * 16;    // 0 / 16
  const int bn   = tid & 63;          // 0..63
  const int bk0  = (tid >> 6) * 8;    // 0,8,16,24

  const int kIters = K >> 5;
  for (int kb = 0; kb < kIters; ++kb) {
    const int kBase = kb * 32;
    __syncthreads();
    // ---- stage A tile [128][32] ----
    if (A_F32) {
      const float* ar = (const float*)Ap + (size_t)(mBase + mrow) * lda + kBase + koff;
      __align__(16) __bf16 tmp[16];
#pragma unroll
      for (int i = 0; i < 16; ++i) tmp[i] = (__bf16)ar[i];
      *(uint4*)&As[mrow][koff]     = *(const uint4*)&tmp[0];
      *(uint4*)&As[mrow][koff + 8] = *(const uint4*)&tmp[8];
    } else {
      // bf16 source: async global->LDS, no VGPR round trip (32B per thread)
      const __bf16* ar = (const __bf16*)Ap + (size_t)(mBase + mrow) * lda + kBase + koff;
      async_copy32(lds_addr_of(&As[mrow][koff]), (unsigned long long)ar);
    }
    // ---- stage B tile transposed [64 n][32 k] ----
    {
      __align__(16) __bf16 tmp[8];
#pragma unroll
      for (int i = 0; i < 8; ++i)
        tmp[i] = (__bf16)Bp[(size_t)(kBase + bk0 + i) * ldb + nBase + bn];
      *(uint4*)&BsT[bn][bk0] = *(const uint4*)&tmp[0];
    }
    if (!A_F32) wait_async0();
    __syncthreads();

    // ---- compute: 4 WMMAs per wave per 32-deep chunk ----
    FragBF a0, a1, b0, b1;
    const int r0 = wm + l15, r1 = wm + 16 + l15;
    a0.q[0] = *(const uint4*)&As[r0][8 * half];
    a0.q[1] = *(const uint4*)&As[r0][8 * half + 16];
    a1.q[0] = *(const uint4*)&As[r1][8 * half];
    a1.q[1] = *(const uint4*)&As[r1][8 * half + 16];
    const int c0 = wn + l15, c1 = wn + 16 + l15;
    b0.q[0] = *(const uint4*)&BsT[c0][8 * half];
    b0.q[1] = *(const uint4*)&BsT[c0][8 * half + 16];
    b1.q[0] = *(const uint4*)&BsT[c1][8 * half];
    b1.q[1] = *(const uint4*)&BsT[c1][8 * half + 16];

    acc[0][0] = wmma_bf16(a0.v, b0.v, acc[0][0]);
    acc[0][1] = wmma_bf16(a0.v, b1.v, acc[0][1]);
    acc[1][0] = wmma_bf16(a1.v, b0.v, acc[1][0]);
    acc[1][1] = wmma_bf16(a1.v, b1.v, acc[1][1]);
  }

  // ---- epilogue: C layout m = i + 8*half, n = lane%16 ----
#pragma unroll
  for (int mt = 0; mt < 2; ++mt)
#pragma unroll
    for (int nt = 0; nt < 2; ++nt)
#pragma unroll
      for (int i = 0; i < 8; ++i) {
        const int m = mBase + wm + mt * 16 + i + 8 * half;
        const int n = nBase + wn + nt * 16 + l15;
        const float val = acc[mt][nt][i];
        if (OUT_BF16) ((__bf16*)Op)[(size_t)m * ldo + n] = (__bf16)val;
        else          ((float*)Op)[(size_t)m * ldo + n] = val;
      }
}

// ---------------------------------------------------------------------------
// Flash attention. Y = packed bf16 [8192][3072] (Q | K | V columns).
// Head-major row qr in [0,16384) per batch maps back through the reference's
// raw reshape: Y row = b*2048 + qr/8, column block = (qr%8)*128.
// ---------------------------------------------------------------------------
static __device__ __forceinline__ const __bf16* yrow(const __bf16* Y, int b,
                                                     int qr, int part) {
  return Y + (size_t)(b * 2048 + (qr >> 3)) * 3072 + part * 1024 + (qr & 7) * 128;
}

__global__ __launch_bounds__(128)
void flash_attn_kernel(const __bf16* __restrict__ Y, __bf16* __restrict__ O) {
  __shared__ __align__(16) __bf16 Kt[32][136];     // K tile, t-major
  __shared__ __align__(16) __bf16 Vt[128][40];     // V tile, c-major (transposed)
  __shared__ __align__(16) __bf16 Ps[4][16][40];   // per-wave P scratch

  const int tid  = threadIdx.x;
  const int lane = tid & 31, wv = tid >> 5;
  const int l15  = lane & 15, half = lane >> 4;
  const int rb = blockIdx.x & 31;          // 64-row block within head
  const int h  = (blockIdx.x >> 5) & 7;    // head
  const int b  = blockIdx.x >> 8;          // batch
  const int row0 = rb * 64 + wv * 16;      // this wave's 16-row tile
  const float SCALE = 0.08838834764831845f;  // 1/sqrt(128)

  // Q: 16 rows x 128 ch as 4 A-fragments (k = (e<8?e:e+8) + 8*half)
  FragBF qf[4];
  {
    const __bf16* qp = yrow(Y, b, h * 2048 + row0 + l15, 0);
#pragma unroll
    for (int kc = 0; kc < 4; ++kc) {
      qf[kc].q[0] = *(const uint4*)(qp + kc * 32 + 8 * half);
      qf[kc].q[1] = *(const uint4*)(qp + kc * 32 + 8 * half + 16);
    }
  }

  v8f accO[8];
#pragma unroll
  for (int nb = 0; nb < 8; ++nb) accO[nb] = vzero8();
  float mrow[8], lrow[8];
#pragma unroll
  for (int i = 0; i < 8; ++i) { mrow[i] = -1e30f; lrow[i] = 0.f; }

  const int stt = tid >> 2;         // staging row 0..31
  const int sc0 = (tid & 3) * 32;   // staging channel chunk

  const int jtMax = rb * 2 + 1;     // causal bound (32-wide tiles)
  for (int jt = 0; jt <= jtMax; ++jt) {
    const int tbase = jt * 32;
    __syncthreads();
    // stage K tile [32][128] row-major: async global->LDS (64B per thread)
    {
      const __bf16* kp = yrow(Y, b, h * 2048 + tbase + stt, 1) + sc0;
      async_copy64(lds_addr_of(&Kt[stt][sc0]), (unsigned long long)kp);
    }
    // stage V tile transposed -> Vt[c][t] (VGPR pass, strided LDS writes)
    {
      const __bf16* vp = yrow(Y, b, h * 2048 + tbase + stt, 2) + sc0;
      __align__(16) __bf16 tmp[32];
      *(uint4*)&tmp[0]  = ((const uint4*)vp)[0];
      *(uint4*)&tmp[8]  = ((const uint4*)vp)[1];
      *(uint4*)&tmp[16] = ((const uint4*)vp)[2];
      *(uint4*)&tmp[24] = ((const uint4*)vp)[3];
#pragma unroll
      for (int i = 0; i < 32; ++i) Vt[sc0 + i][stt] = tmp[i];
    }
    // prefetch next iteration's K/V rows into cache (global_prefetch_b8)
    if (jt < jtMax) {
      __builtin_prefetch(yrow(Y, b, h * 2048 + tbase + 32 + stt, 1) + sc0, 0, 1);
      __builtin_prefetch(yrow(Y, b, h * 2048 + tbase + 32 + stt, 2) + sc0, 0, 1);
    }
    wait_async0();
    __syncthreads();

    // S = Q*K^T: two 16x16 score tiles over 4 k-chunks
    v8f s0 = vzero8(), s1 = vzero8();
#pragma unroll
    for (int kc = 0; kc < 4; ++kc) {
      FragBF kb0, kb1;
      const __bf16* k0p = &Kt[l15][kc * 32 + 8 * half];
      const __bf16* k1p = &Kt[16 + l15][kc * 32 + 8 * half];
      kb0.q[0] = *(const uint4*)k0p;  kb0.q[1] = *(const uint4*)(k0p + 16);
      kb1.q[0] = *(const uint4*)k1p;  kb1.q[1] = *(const uint4*)(k1p + 16);
      s0 = wmma_bf16(qf[kc].v, kb0.v, s0);
      s1 = wmma_bf16(qf[kc].v, kb1.v, s1);
    }

    // online softmax (C layout: row = i + 8*half, col = lane%16)
#pragma unroll
    for (int i = 0; i < 8; ++i) {
      const int r  = row0 + i + 8 * half;
      const int t0 = tbase + l15;
      const int t1 = t0 + 16;
      float a0 = (t0 <= r) ? s0[i] * SCALE : -1e30f;
      float a1 = (t1 <= r) ? s1[i] * SCALE : -1e30f;
      float tm = redmax16(fmaxf(a0, a1));
      float mo = mrow[i];
      float mn = fmaxf(mo, tm);
      float alpha = __expf(mo - mn);
      mrow[i] = mn;
      float p0 = __expf(a0 - mn);
      float p1 = __expf(a1 - mn);
      lrow[i] = lrow[i] * alpha + redsum16(p0 + p1);
#pragma unroll
      for (int nb = 0; nb < 8; ++nb) accO[nb][i] = accO[nb][i] * alpha;
      Ps[wv][i + 8 * half][l15]      = (__bf16)p0;
      Ps[wv][i + 8 * half][16 + l15] = (__bf16)p1;
    }

    // per-wave LDS bounce: DS ops are in-order within a wave; just block
    // compiler reordering and wait the counter.
    asm volatile("s_wait_dscnt 0" ::: "memory");

    // P as a single 16x32 A-fragment
    FragBF pf;
    {
      const __bf16* pp = &Ps[wv][l15][8 * half];
      pf.q[0] = *(const uint4*)pp;
      pf.q[1] = *(const uint4*)(pp + 16);
    }
    // O += P * V  (8 WMMAs across the 128 output channels)
#pragma unroll
    for (int nb = 0; nb < 8; ++nb) {
      FragBF vb;
      const __bf16* vp2 = &Vt[nb * 16 + l15][8 * half];
      vb.q[0] = *(const uint4*)vp2;
      vb.q[1] = *(const uint4*)(vp2 + 16);
      accO[nb] = wmma_bf16(pf.v, vb.v, accO[nb]);
    }
  }

  // normalize + store bf16 O[b][qr][c]
#pragma unroll
  for (int i = 0; i < 8; ++i) {
    const float inv = 1.0f / lrow[i];
    const int r = row0 + i + 8 * half;
    __bf16* op = O + ((size_t)b * 16384 + (size_t)h * 2048 + r) * 128;
#pragma unroll
    for (int nb = 0; nb < 8; ++nb)
      op[nb * 16 + l15] = (__bf16)(accO[nb][i] * inv);
  }
}

// ---------------------------------------------------------------------------
extern "C" void kernel_launch(void* const* d_in, const int* in_sizes, int n_in,
                              void* d_out, int out_size, void* d_ws, size_t ws_size,
                              hipStream_t stream) {
  (void)in_sizes; (void)n_in; (void)out_size; (void)ws_size;
  const float* x    = (const float*)d_in[0];   // [4,2048,128]
  const float* qkv  = (const float*)d_in[1];   // [128,3072]
  const float* proj = (const float*)d_in[2];   // [1024,128]
  float* out = (float*)d_out;                  // [4,2048,128]

  __bf16* Yb = (__bf16*)d_ws;                                    // [8192][3072] bf16
  __bf16* Ob = (__bf16*)((char*)d_ws + (size_t)8192 * 3072 * 2); // [4][16384][128] bf16

  // K1: Y = x @ qkv   (M=8192, N=3072, K=128)
  wmma_gemm_kernel<true, true><<<dim3(64, 48), 256, 0, stream>>>(
      (const void*)x, qkv, (void*)Yb, 128, 128, 3072, 3072);

  // K2: flash attention, blockIdx = b*256 + h*32 + rb
  flash_attn_kernel<<<dim3(1024), 128, 0, stream>>>(Yb, Ob);

  // K3: out = O @ projection  (M=8192, N=128, K=1024)
  wmma_gemm_kernel<false, false><<<dim3(64, 2), 256, 0, stream>>>(
      (const void*)Ob, proj, (void*)out, 1024, 1024, 128, 128);
}
